// Seq2Seq_27101243637848
// MI455X (gfx1250) — matrix operand order repaired and checked by reading, not compiled, not run on hardware
//
#include <hip/hip_runtime.h>

// ---------------------------------------------------------------------------
// Seq2Seq (embed -> LSTM encoder -> LSTM decoder -> FC logits) for gfx1250.
// Strategy:
//   * All GEMMs use v_wmma_f32_16x16x32_f16 (f16 in, fp32 accumulate).
//   * LSTM recurrence: batch rows are independent -> 4 persistent workgroups
//     (16 rows each), h kept in LDS (f16), c kept in registers in the WMMA
//     C/D fragment layout, only __syncthreads between time steps.
//   * xg = x@W_ih^T + b_ih + b_hh precomputed once, stored transposed so the
//     LSTM loads it directly as the WMMA C-fragment (32B/lane coalesced).
//   * FC GEMM [4032,256]x[256,32000]: double-buffered LDS B tiles filled via
//     global_load_async_to_lds_b128 (ASYNCcnt), overlapped with WMMA.
// ---------------------------------------------------------------------------

typedef __attribute__((ext_vector_type(16))) _Float16 v16h;
typedef __attribute__((ext_vector_type(8)))  float    v8f;

union ABFrag { v16h v; uint4 q[2]; };

__device__ __forceinline__ v8f wmma_f16(v16h a, v16h b, v8f c) {
  return __builtin_amdgcn_wmma_f32_16x16x32_f16(
      /*neg_a=*/false, a, /*neg_b=*/false, b,
      /*c_mod=*/(short)0, c, /*reuse_a=*/false, /*reuse_b=*/false);
}

__device__ __forceinline__ float sigmoidf_(float x) {
  return 1.0f / (1.0f + __expf(-x));
}

// Async copy of 32 bytes global -> LDS (two B128 beats; the instruction
// offset is added to BOTH the global and the LDS address per CDNA5 ISA).
__device__ __forceinline__ void async_copy32(uint32_t lds_off,
                                             const void* gsrc) {
  asm volatile(
      "global_load_async_to_lds_b128 %0, %1, off\n\t"
      "global_load_async_to_lds_b128 %0, %1, off offset:16"
      :
      : "v"(lds_off), "v"(gsrc)
      : "memory");
}

__device__ __forceinline__ void wait_async0() {
  asm volatile("s_wait_asynccnt 0" ::: "memory");
}

// ---------------------------------------------------------------------------
// f32 -> f16 conversion
// ---------------------------------------------------------------------------
__global__ void cvt_f16_kernel(const float* __restrict__ src,
                               _Float16* __restrict__ dst, int n) {
  int i = blockIdx.x * blockDim.x + threadIdx.x;
  if (i < n) dst[i] = (_Float16)src[i];
}

// ---------------------------------------------------------------------------
// Embedding gather + concat -> X[(t*64+b)*192 + d] (f16)
// token/pos/ner index arrays are [64][Tstride]; we use column t (t < steps).
// ---------------------------------------------------------------------------
__global__ void gather_kernel(const int* __restrict__ tok,
                              const int* __restrict__ pos,
                              const int* __restrict__ ner,
                              const float* __restrict__ tok_emb,
                              const float* __restrict__ pos_emb,
                              const float* __restrict__ ner_emb,
                              _Float16* __restrict__ X,
                              int steps, int Tstride) {
  int idx = blockIdx.x * blockDim.x + threadIdx.x;
  int total = steps * 64 * 192;
  if (idx >= total) return;
  int d = idx % 192;
  int m = idx / 192;
  int b = m & 63;
  int t = m >> 6;
  float v;
  if (d < 128)      v = tok_emb[(size_t)tok[b * Tstride + t] * 128 + d];
  else if (d < 160) v = pos_emb[(size_t)pos[b * Tstride + t] * 32 + (d - 128)];
  else              v = ner_emb[(size_t)ner[b * Tstride + t] * 32 + (d - 160)];
  X[idx] = (_Float16)v;
}

// ---------------------------------------------------------------------------
// xg = X @ W_ih^T + (b_ih + b_hh), stored transposed:
//   xgT[g][t][j][mlocal]  with m = t*64 + (16*g + mlocal), j in [0,1024)
// One wave per (mtile, jtile); K = 192 = 6 chunks of 32.
// ---------------------------------------------------------------------------
__global__ __launch_bounds__(256)
void xg_gemm_kernel(const _Float16* __restrict__ X,    // [steps*64][192]
                    const _Float16* __restrict__ W16,  // [1024][192]
                    const float* __restrict__ b_ih,
                    const float* __restrict__ b_hh,
                    float* __restrict__ xgT,           // [4][steps][1024][16]
                    int steps) {
  int wid = blockIdx.x * (blockDim.x >> 5) + (threadIdx.x >> 5);
  int mtiles = steps * 4;
  if (wid >= mtiles * 64) return;
  int mtile = wid >> 6;   // / 64 jtiles
  int jt    = wid & 63;
  int lane  = threadIdx.x & 31;
  int s = mtile >> 2;
  int g = mtile & 3;
  int jl = jt * 16 + (lane & 15);
  int mrow = mtile * 16 + (lane & 15);
  int grp8 = (lane >> 4) << 3;   // 0 or 8

  float bias = b_ih[jl] + b_hh[jl];
  v8f acc = {bias, bias, bias, bias, bias, bias, bias, bias};

#pragma unroll
  for (int kc = 0; kc < 6; ++kc) {
    ABFrag a;
    const uint4* ap = (const uint4*)(X + (size_t)mrow * 192 + kc * 32 + ((lane >> 4) << 4));
    a.q[0] = ap[0];
    a.q[1] = ap[1];   // +8 halves
    ABFrag b;
    const uint4* bp =
        (const uint4*)(W16 + (size_t)jl * 192 + kc * 32 + ((lane >> 4) << 4));
    b.q[0] = bp[0];
    b.q[1] = bp[1];
    acc = wmma_f16(a.v, b.v, acc);
  }
  // store: per lane j fixed, 8 consecutive mlocal values (32B contiguous)
  float* op = xgT + (((size_t)g * steps + s) * 1024 + jl) * 16 + grp8;
  ((float4*)op)[0] = float4{acc[0], acc[1], acc[2], acc[3]};
  ((float4*)op)[1] = float4{acc[4], acc[5], acc[6], acc[7]};
}

// ---------------------------------------------------------------------------
// Persistent per-batch-group LSTM. grid = 4 blocks x 512 threads (16 waves).
// Wave w owns gate-unit columns u in [16w,16w+16) for all 16 batch rows;
// its 4 tiles (i,f,g,o) live at j = gate*256 + 16w.  c stays in registers.
// ---------------------------------------------------------------------------
__global__ __launch_bounds__(512)
void lstm_kernel(const float* __restrict__ xgT,       // [4][steps][1024][16]
                 const _Float16* __restrict__ Whh16,  // [1024][256]
                 const float* __restrict__ c_init,    // [64][256]-ish or null
                 const _Float16* __restrict__ h_init, // [64][256] or null
                 float* __restrict__ c_out,           // or null
                 _Float16* __restrict__ h_out,        // or null
                 _Float16* __restrict__ hs_out,       // [steps][64][256] or null
                 int steps) {
  __shared__ __align__(16) _Float16 hbuf[16 * 256];  // 8KB, row-major [m][k]

  const int g    = blockIdx.x;
  const int tid  = threadIdx.x;
  const int w    = tid >> 5;
  const int lane = tid & 31;
  const int nlo  = lane & 15;
  const int grp8 = (lane >> 4) << 3;

  // init h in LDS
  if (h_init) {
    ((uint4*)hbuf)[tid] = ((const uint4*)(h_init + (size_t)g * 16 * 256))[tid];
  } else {
    ((uint4*)hbuf)[tid] = make_uint4(0u, 0u, 0u, 0u);
  }
  // init c in registers (fragment layout)
  v8f c = {0.f, 0.f, 0.f, 0.f, 0.f, 0.f, 0.f, 0.f};
  if (c_init) {
    const float4* cp = (const float4*)(c_init + (((size_t)g * 16 + w) * 32 + lane) * 8);
    float4 c0 = cp[0], c1 = cp[1];
    c[0] = c0.x; c[1] = c0.y; c[2] = c0.z; c[3] = c0.w;
    c[4] = c1.x; c[5] = c1.y; c[6] = c1.z; c[7] = c1.w;
  }
  __syncthreads();

  const float* xg_g = xgT + (size_t)g * steps * 1024 * 16;

  for (int t = 0; t < steps; ++t) {
    // A fragments for all 8 K-chunks (h is shared across this wave's 4 tiles)
    ABFrag a[8];
    const uint4* hp = (const uint4*)(hbuf + nlo * 256);
#pragma unroll
    for (int kc = 0; kc < 8; ++kc) {
      a[kc].q[0] = hp[(kc * 32 + ((lane >> 4) << 4)) >> 3];
      a[kc].q[1] = hp[(kc * 32 + ((lane >> 4) << 4) + 8) >> 3];
    }

    const float* xg_t = xg_g + (size_t)t * 1024 * 16;
    v8f d[4];
#pragma unroll
    for (int gi = 0; gi < 4; ++gi) {
      int j0 = gi * 256 + w * 16;
      int jl = j0 + nlo;
      // C init = xg tile (biases already folded), 32B/lane coalesced
      const float4* cxp = (const float4*)(xg_t + (size_t)jl * 16 + grp8);
      float4 x0 = cxp[0], x1 = cxp[1];
      v8f acc = {x0.x, x0.y, x0.z, x0.w, x1.x, x1.y, x1.z, x1.w};
#pragma unroll
      for (int kc = 0; kc < 8; ++kc) {
        ABFrag b;
        const uint4* bp = (const uint4*)(Whh16 + (size_t)jl * 256 + kc * 32 +
                                         ((lane >> 4) << 4));
        b.q[0] = bp[0];
        b.q[1] = bp[1];
        acc = wmma_f16(a[kc].v, b.v, acc);
      }
      d[gi] = acc;
    }

    __syncthreads();  // everyone done reading hbuf

    // elementwise cell update; write h (f16) back to LDS (+ optional hs)
#pragma unroll
    for (int r = 0; r < 8; ++r) {
      float iv = sigmoidf_(d[0][r]);
      float fv = sigmoidf_(d[1][r]);
      float gv = tanhf(d[2][r]);
      float ov = sigmoidf_(d[3][r]);
      float cr = fv * c[r] + iv * gv;
      c[r] = cr;
      float hv = ov * tanhf(cr);
      int m = r + grp8;
      int u = w * 16 + nlo;
      _Float16 h16 = (_Float16)hv;
      hbuf[m * 256 + u] = h16;
      if (hs_out)
        hs_out[((size_t)t * 64 + g * 16 + m) * 256 + u] = h16;
    }
    __syncthreads();  // hbuf ready for next step
  }

  if (c_out) {
    float4* cp = (float4*)(c_out + (((size_t)g * 16 + w) * 32 + lane) * 8);
    cp[0] = float4{c[0], c[1], c[2], c[3]};
    cp[1] = float4{c[4], c[5], c[6], c[7]};
  }
  if (h_out) {
    ((uint4*)(h_out + (size_t)g * 16 * 256))[tid] = ((const uint4*)hbuf)[tid];
  }
}

// ---------------------------------------------------------------------------
// FC: out[b][t][v] = hs[m=t*64+b] . fc_W[v] + fc_b[v]
// grid = (250 N-groups, 63 M-groups), block = 128 (4 waves).
// Each wave: 1 M-tile x 8 N-tiles.  B chunks are staged into double-buffered
// LDS with global_load_async_to_lds_b128 and overlapped with WMMA compute.
// ---------------------------------------------------------------------------
__global__ __launch_bounds__(128)
void fc_kernel(const _Float16* __restrict__ hs,   // [4032][256]
               const _Float16* __restrict__ W16,  // [32000][256]
               const float* __restrict__ bias,    // [32000]
               float* __restrict__ out) {         // [64][63][32000]
  __shared__ __align__(16) _Float16 btile[2][128 * 32];  // 2 x 8KB

  const int ng   = blockIdx.x;        // 0..249
  const int mg   = blockIdx.y;        // 0..62
  const int w    = threadIdx.x >> 5;  // 0..3
  const int lane = threadIdx.x & 31;
  const int nlo  = lane & 15;
  const int grp8 = (lane >> 4) << 3;

  const int mtile = mg * 4 + w;
  const int m0    = mtile * 16;
  const int mrow  = m0 + nlo;
  const int v0    = ng * 128;

  v8f acc[8];
#pragma unroll
  for (int nt = 0; nt < 8; ++nt) {
    float bb = bias[v0 + nt * 16 + nlo];
    acc[nt] = v8f{bb, bb, bb, bb, bb, bb, bb, bb};
  }

  // Each thread owns one 64-byte row-chunk of the B panel per K-chunk.
  const int vr = threadIdx.x;  // 0..127 -> row v0+vr
  const _Float16* gbase = W16 + (size_t)(v0 + vr) * 256;

  // Prologue: issue async fill of buffer 0 (chunk 0).
  async_copy32((uint32_t)(uintptr_t)(&btile[0][vr * 32]), gbase + 0 * 32);

  for (int kc = 0; kc < 8; ++kc) {
    const int cur = kc & 1;
    wait_async0();     // my async writes (chunk kc) are in LDS
    __syncthreads();   // everyone's writes visible; prev-buffer reads done

    // Issue next chunk into the other buffer while we compute this one.
    if (kc < 7) {
      async_copy32((uint32_t)(uintptr_t)(&btile[cur ^ 1][vr * 32]),
                   gbase + (kc + 1) * 32);
    }

    ABFrag a;
    const uint4* ap = (const uint4*)(hs + (size_t)mrow * 256 + kc * 32 + ((lane >> 4) << 4));
    a.q[0] = ap[0];
    a.q[1] = ap[1];
#pragma unroll
    for (int nt = 0; nt < 8; ++nt) {
      ABFrag b;
      const uint4* bp = (const uint4*)(&btile[cur][0] + (nt * 16 + nlo) * 32 +
                                       ((lane >> 4) << 4));
      b.q[0] = bp[0];
      b.q[1] = bp[1];
      acc[nt] = wmma_f16(a.v, b.v, acc[nt]);
    }
  }

#pragma unroll
  for (int nt = 0; nt < 8; ++nt) {
    int v = v0 + nt * 16 + nlo;
#pragma unroll
    for (int r = 0; r < 8; ++r) {
      int m = m0 + r + grp8;
      int t = m >> 6;
      int b = m & 63;
      out[((size_t)b * 63 + t) * 32000 + v] = acc[nt][r];
    }
  }
}

// ---------------------------------------------------------------------------
// Host launcher
// ---------------------------------------------------------------------------
extern "C" void kernel_launch(void* const* d_in, const int* in_sizes, int n_in,
                              void* d_out, int out_size, void* d_ws,
                              size_t ws_size, hipStream_t stream) {
  (void)in_sizes; (void)n_in; (void)out_size; (void)ws_size;

  const int* src_token = (const int*)d_in[0];
  const int* src_pos   = (const int*)d_in[1];
  const int* src_ner   = (const int*)d_in[2];
  const int* trg_token = (const int*)d_in[3];
  const int* trg_pos   = (const int*)d_in[4];
  const int* trg_ner   = (const int*)d_in[5];
  const float* enc_tok_emb = (const float*)d_in[6];
  const float* enc_pos_emb = (const float*)d_in[7];
  const float* enc_ner_emb = (const float*)d_in[8];
  const float* enc_W_ih = (const float*)d_in[9];
  const float* enc_W_hh = (const float*)d_in[10];
  const float* enc_b_ih = (const float*)d_in[11];
  const float* enc_b_hh = (const float*)d_in[12];
  const float* dec_tok_emb = (const float*)d_in[13];
  const float* dec_pos_emb = (const float*)d_in[14];
  const float* dec_ner_emb = (const float*)d_in[15];
  const float* dec_W_ih = (const float*)d_in[16];
  const float* dec_W_hh = (const float*)d_in[17];
  const float* dec_b_ih = (const float*)d_in[18];
  const float* dec_b_hh = (const float*)d_in[19];
  const float* fc_W = (const float*)d_in[20];
  const float* fc_b = (const float*)d_in[21];
  float* out = (float*)d_out;

  // ---- workspace carve-up (256B aligned) ----
  char* ws = (char*)d_ws;
  size_t off = 0;
  auto take = [&](size_t bytes) -> char* {
    char* p = ws + off;
    off += (bytes + 255) & ~(size_t)255;
    return p;
  };
  _Float16* wih_e16 = (_Float16*)take((size_t)1024 * 192 * 2);
  _Float16* whh_e16 = (_Float16*)take((size_t)1024 * 256 * 2);
  _Float16* wih_d16 = (_Float16*)take((size_t)1024 * 192 * 2);
  _Float16* whh_d16 = (_Float16*)take((size_t)1024 * 256 * 2);
  _Float16* fcw16   = (_Float16*)take((size_t)32000 * 256 * 2);
  _Float16* x_enc   = (_Float16*)take((size_t)4096 * 192 * 2);
  _Float16* x_dec   = (_Float16*)take((size_t)4032 * 192 * 2);
  float*    xgT_e   = (float*)take((size_t)4 * 64 * 1024 * 16 * 4);
  float*    xgT_d   = (float*)take((size_t)4 * 63 * 1024 * 16 * 4);
  float*    c_buf   = (float*)take((size_t)64 * 256 * 4);
  _Float16* h_buf   = (_Float16*)take((size_t)64 * 256 * 2);
  _Float16* hs16    = (_Float16*)take((size_t)4032 * 256 * 2);

  // ---- weight conversions to f16 ----
  auto cvt = [&](const float* s, _Float16* d, int n) {
    cvt_f16_kernel<<<(n + 255) / 256, 256, 0, stream>>>(s, d, n);
  };
  cvt(enc_W_ih, wih_e16, 1024 * 192);
  cvt(enc_W_hh, whh_e16, 1024 * 256);
  cvt(dec_W_ih, wih_d16, 1024 * 192);
  cvt(dec_W_hh, whh_d16, 1024 * 256);
  cvt(fc_W, fcw16, 32000 * 256);

  // ---- embedding gather (encoder: 64 steps, decoder: 63 steps) ----
  gather_kernel<<<(64 * 64 * 192) / 256, 256, 0, stream>>>(
      src_token, src_pos, src_ner, enc_tok_emb, enc_pos_emb, enc_ner_emb,
      x_enc, 64, 64);
  gather_kernel<<<(63 * 64 * 192) / 256, 256, 0, stream>>>(
      trg_token, trg_pos, trg_ner, dec_tok_emb, dec_pos_emb, dec_ner_emb,
      x_dec, 63, 64);

  // ---- xg precompute (biases folded, transposed layout) ----
  xg_gemm_kernel<<<(64 * 4 * 64) / 8, 256, 0, stream>>>(
      x_enc, wih_e16, enc_b_ih, enc_b_hh, xgT_e, 64);
  xg_gemm_kernel<<<(63 * 4 * 64) / 8, 256, 0, stream>>>(
      x_dec, wih_d16, dec_b_ih, dec_b_hh, xgT_d, 63);

  // ---- encoder LSTM (h0=c0=0) -> final (h,c) ----
  lstm_kernel<<<4, 512, 0, stream>>>(xgT_e, whh_e16, nullptr, nullptr,
                                     c_buf, h_buf, nullptr, 64);
  // ---- decoder LSTM (init from encoder) -> hs (f16) ----
  lstm_kernel<<<4, 512, 0, stream>>>(xgT_d, whh_d16, c_buf, h_buf,
                                     nullptr, nullptr, hs16, 63);

  // ---- output projection ----
  fc_kernel<<<dim3(250, 63), 128, 0, stream>>>(hs16, fcw16, fc_b, out);
}
